// MyLinkPredictionGCN_25013889532262
// MI455X (gfx1250) — matrix operand order repaired and checked by reading, not compiled, not run on hardware
//
#include <hip/hip_runtime.h>
#include <hip/hip_bf16.h>
#include <math.h>

// ---------------- problem constants (from reference) ----------------
#define N_NODES 100000
#define NNZ_    1000000
#define F_IN    1433
#define F_HID   128
#define F_OUT   64
#define NPOS    20000
#define NNEG    20000
#define EPSL    1e-6f

// GEMM blocking: 10 waves/block (6250 row-tiles = 625 blocks exactly),
// B staged through LDS in chunks of 4 k-steps.
#define GEMM_WAVES  10
#define GEMM_BLOCK  (GEMM_WAVES * 32)
#define CHUNK_STEPS 4

// ---------------- CDNA5 WMMA vector types ----------------
typedef __attribute__((ext_vector_type(16))) __bf16 v16bf;
typedef __attribute__((ext_vector_type(8)))  float  v8f;

// f32 -> bf16 round-to-nearest-even via integer path.
static __device__ __forceinline__ __bf16 f2bf(float f) {
  union { float f; unsigned u; } v; v.f = f;
  unsigned r = v.u + 0x7fffu + ((v.u >> 16) & 1u);
  union { unsigned short s; __bf16 b; } o;
  o.s = (unsigned short)(r >> 16);
  return o.b;
}

// =====================================================================
// 1) Pack W[K][F] (f32 row-major) -> bf16 WMMA B fragments.
//    Entry t = (s*colTiles + ct)*32 + lane holds lane's 16 bf16:
//    n = ct*16 + (lane&15),  k = s*32 + 16*(lane>>4) + i.
// =====================================================================
__global__ void gcn_pack_w(const float* __restrict__ W, v16bf* __restrict__ Wp,
                           int K, int F) {
  int t = blockIdx.x * blockDim.x + threadIdx.x;
  int ksteps = (K + 31) >> 5;
  int colTiles = F >> 4;
  int total = colTiles * ksteps * 32;
  if (t >= total) return;
  int lane = t & 31;
  int g = t >> 5;
  int ct = g % colTiles;
  int s  = g / colTiles;
  int half = lane >> 4;
  int n = ct * 16 + (lane & 15);
  int kb = s * 32 + half * 16;
  v16bf b;
#pragma unroll
  for (int i = 0; i < 16; ++i) {
    int k = kb + i;
    b[i] = (k < K) ? f2bf(W[(long)k * F + n]) : f2bf(0.f);
  }
  Wp[t] = b;
}

// =====================================================================
// 2) GEMM: Y[N][F] = X[N][K] @ W.  One wave32 per 16-row strip, all F
//    columns (CT = F/16 accumulators) -> A read from HBM exactly once.
//    All 10 waves of a block share B: staged via LDS in 4-step chunks
//    (coalesced b128 global loads -> ds_store_b128; fragments re-read
//    from LDS instead of hammering L2 once per wave).
// =====================================================================
template <int CT>
__global__ __launch_bounds__(GEMM_BLOCK) void gcn_gemm_wmma(
    const float* __restrict__ X, const v16bf* __restrict__ Wp,
    float* __restrict__ Y, int N, int K) {
  constexpr int F = CT * 16;
  __shared__ v16bf ldsW[CHUNK_STEPS * CT * 32];    // 32KB (CT=8) / 16KB (CT=4)

  const int lane = threadIdx.x & 31;
  const int wave = threadIdx.x >> 5;
  const int numRowTiles = N >> 4;
  int rowTile = blockIdx.x * GEMM_WAVES + wave;
  if (rowTile >= numRowTiles) rowTile = numRowTiles - 1;  // keep barriers uniform
  const int half = lane >> 4;
  const long row = (long)rowTile * 16 + (lane & 15);
  const int kfull  = K >> 5;                 // steps with no K-guard needed
  const int ksteps = (K + 31) >> 5;

  const float* xrow = X + row * K;

  v8f acc[CT];
#pragma unroll
  for (int c = 0; c < CT; ++c) acc[c] = v8f{};

  for (int cs = 0; cs < ksteps; cs += CHUNK_STEPS) {
    const int nsteps = (ksteps - cs < CHUNK_STEPS) ? (ksteps - cs) : CHUNK_STEPS;

    // ---- cooperative B stage: global(b128) -> LDS(b128), coalesced ----
    {
      const uint4* src = (const uint4*)(Wp + (long)cs * (CT * 32));
      uint4* dst = (uint4*)ldsW;
      const int n16 = nsteps * CT * 64;      // 16B units in this chunk
      for (int u = threadIdx.x; u < n16; u += GEMM_BLOCK) dst[u] = src[u];
    }
    __syncthreads();

    // ---- compute nsteps k-steps with B from LDS ----
    for (int sl = 0; sl < nsteps; ++sl) {
      const int s = cs + sl;
      const int kb = s * 32 + half * 16;
      v16bf a;
      if (s < kfull) {                       // wave-uniform branch
        __builtin_prefetch(xrow + kb + 160, 0, 1);   // global_prefetch on A stream
        float a0[8], a1[8];
        __builtin_memcpy(a0, xrow + kb, 32);         // two contiguous 32B runs
        __builtin_memcpy(a1, xrow + kb + 8, 32);
#pragma unroll
        for (int i = 0; i < 8; ++i) {
          a[i]     = f2bf(a0[i]);
          a[i + 8] = f2bf(a1[i]);
        }
      } else {                               // K tail: clamp + cndmask, no branch
#pragma unroll
        for (int i = 0; i < 8; ++i) {
          int k0 = kb + i, k1 = kb + 8 + i;
          float v0 = xrow[k0 < K ? k0 : K - 1];
          float v1 = xrow[k1 < K ? k1 : K - 1];
          a[i]     = f2bf(k0 < K ? v0 : 0.f);
          a[i + 8] = f2bf(k1 < K ? v1 : 0.f);
        }
      }
      const v16bf* wrow = ldsW + sl * (CT * 32);
#pragma unroll
      for (int c = 0; c < CT; ++c) {
        v16bf b = wrow[c * 32 + lane];       // 32B LDS fragment (ds_load_b128 x2)
        acc[c] = __builtin_amdgcn_wmma_f32_16x16x32_bf16(
            false, a, false, b, (short)0, acc[c], false, false);
      }
    }
    __syncthreads();
  }

  // D layout: lane col n = lane&15, element j -> row j + 8*half
  const int n = lane & 15;
#pragma unroll
  for (int c = 0; c < CT; ++c) {
#pragma unroll
    for (int j = 0; j < 8; ++j) {
      long mrow = (long)rowTile * 16 + j + 8 * half;
      Y[mrow * F + c * 16 + n] = acc[c][j];
    }
  }
}

// =====================================================================
// 3) h[r][f] = b[f]   (pre-fill bias; SpMM accumulates on top)
// =====================================================================
__global__ void gcn_fill_bias(float* __restrict__ y, const float* __restrict__ b,
                              long total, int fmask) {
  long idx = (long)blockIdx.x * blockDim.x + threadIdx.x;
  if (idx >= total) return;
  y[idx] = b[idx & fmask];
}

__global__ void gcn_zero(float* __restrict__ p, int n) {
  int i = blockIdx.x * blockDim.x + threadIdx.x;
  if (i < n) p[i] = 0.f;
}

// =====================================================================
// 4) SpMM scatter: one wave32 per nnz.  y[row[e]][:] += val[e]*x[col[e]][:]
//    (unsorted segment_sum -> global_atomic_add_f32; y is L2-resident)
// =====================================================================
__global__ __launch_bounds__(256) void gcn_spmm_atomic(
    const int* __restrict__ rows, const int* __restrict__ cols,
    const float* __restrict__ vals, const float* __restrict__ x,
    float* __restrict__ y, int nnz, int F) {
  int e = (int)(((long)blockIdx.x * blockDim.x + threadIdx.x) >> 5);
  int lane = threadIdx.x & 31;
  if (e >= nnz) return;
  const float v = vals[e];
  const float* xs = x + (long)cols[e] * F;
  float* yd = y + (long)rows[e] * F;
  for (int f = lane; f < F; f += 32)
    atomicAdd(&yd[f], v * xs[f]);
}

// =====================================================================
// 5) In-place ReLU + column-sum (register accumulation, 1 atomic/col/block)
// =====================================================================
__global__ void gcn_relu_colsum(float* __restrict__ h, float* __restrict__ colsum,
                                int n, int F, int rowsPerBlock) {
  int f = threadIdx.x;                       // blockDim.x == F
  int r0 = blockIdx.x * rowsPerBlock;
  float acc = 0.f;
  for (int r = 0; r < rowsPerBlock; ++r) {
    int row = r0 + r;
    if (row >= n) break;
    long idx = (long)row * F + f;
    float v = h[idx];
    v = v > 0.f ? v : 0.f;
    h[idx] = v;
    acc += v;
  }
  atomicAdd(&colsum[f], acc);
}

// =====================================================================
// 6) PairNorm (PN-SI): x -= colmean; x /= sqrt(eps + ||x_row||^2)
//    One block per row; shuffle reduce per wave + LDS cross-wave combine.
// =====================================================================
__global__ void gcn_pairnorm(float* __restrict__ h, const float* __restrict__ colsum,
                             int n, int F) {
  __shared__ float red[4];
  int f = threadIdx.x;                       // blockDim.x == F (128 or 64)
  int lane = f & 31, wv = f >> 5;
  long idx = (long)blockIdx.x * F + f;
  float v = h[idx] - colsum[f] * (1.0f / (float)n);
  float ss = v * v;
#pragma unroll
  for (int off = 16; off; off >>= 1) ss += __shfl_xor(ss, off, 32);
  if (lane == 0) red[wv] = ss;
  __syncthreads();
  float tot = 0.f;
  int nw = F >> 5;
  for (int w = 0; w < nw; ++w) tot += red[w];
  h[idx] = v * __frsqrt_rn(EPSL + tot);
}

// =====================================================================
// 7) Decode: sigmoid(dot(x[a], x[b])) — one wave32 per edge, shuffle reduce
// =====================================================================
__global__ __launch_bounds__(256) void gcn_decode(
    const float* __restrict__ X, const int* __restrict__ pos,
    const int* __restrict__ neg, float* __restrict__ out,
    int nPos, int nNeg, int F) {
  int wv = (int)((long)blockIdx.x * blockDim.x + threadIdx.x) >> 5;
  int lane = threadIdx.x & 31;
  int E = nPos + nNeg;
  if (wv >= E) return;
  const int* ei = (wv < nPos) ? (pos + 2 * (long)wv) : (neg + 2 * (long)(wv - nPos));
  const float* xa = X + (long)ei[0] * F;
  const float* xb = X + (long)ei[1] * F;
  float s = 0.f;
  for (int f = lane; f < F; f += 32) s += xa[f] * xb[f];
#pragma unroll
  for (int off = 16; off; off >>= 1) s += __shfl_xor(s, off, 32);
  if (lane == 0) out[wv] = 1.f / (1.f + __expf(-s));
}

// =====================================================================
// Host orchestration (all launches on `stream`; graph-capture safe)
// =====================================================================
static void run_layer(const float* xin, int K, const float* W, const float* b, int F,
                      const int* arow, const int* acol, const float* aval,
                      float* sup, float* hout, v16bf* Wp, float* colsum,
                      hipStream_t stream) {
  int ksteps = (K + 31) / 32;
  int colTiles = F / 16;
  int packTotal = colTiles * ksteps * 32;
  gcn_pack_w<<<(packTotal + 255) / 256, 256, 0, stream>>>(W, Wp, K, F);

  int blocks = (N_NODES / 16 + GEMM_WAVES - 1) / GEMM_WAVES;   // 625 exact
  if (F == 128)
    gcn_gemm_wmma<8><<<blocks, GEMM_BLOCK, 0, stream>>>(xin, Wp, sup, N_NODES, K);
  else
    gcn_gemm_wmma<4><<<blocks, GEMM_BLOCK, 0, stream>>>(xin, Wp, sup, N_NODES, K);

  long totalNF = (long)N_NODES * F;
  gcn_fill_bias<<<(int)((totalNF + 255) / 256), 256, 0, stream>>>(hout, b, totalNF, F - 1);
  gcn_zero<<<1, F, 0, stream>>>(colsum, F);

  gcn_spmm_atomic<<<NNZ_ / 8, 256, 0, stream>>>(arow, acol, aval, sup, hout, NNZ_, F);

  gcn_relu_colsum<<<N_NODES / 100, F, 0, stream>>>(hout, colsum, N_NODES, F, 100);
  gcn_pairnorm<<<N_NODES, F, 0, stream>>>(hout, colsum, N_NODES, F);
}

extern "C" void kernel_launch(void* const* d_in, const int* in_sizes, int n_in,
                              void* d_out, int out_size, void* d_ws, size_t ws_size,
                              hipStream_t stream) {
  const float* X    = (const float*)d_in[0];
  const int*   arow = (const int*)d_in[1];
  const int*   acol = (const int*)d_in[2];
  const float* aval = (const float*)d_in[3];
  const int*   pos  = (const int*)d_in[4];
  const int*   neg  = (const int*)d_in[5];
  const float* W0   = (const float*)d_in[6];
  const float* b0   = (const float*)d_in[7];
  const float* W1   = (const float*)d_in[8];
  const float* b1   = (const float*)d_in[9];
  const float* W2   = (const float*)d_in[10];
  const float* b2   = (const float*)d_in[11];
  float* out = (float*)d_out;

  // Workspace layout (~104 MB): bufA | bufB | packed-W (1 MB) | colsum
  char* ws = (char*)d_ws;
  size_t bufBytes = (size_t)N_NODES * F_HID * sizeof(float);   // 51.2 MB
  float* bufA   = (float*)ws;
  float* bufB   = (float*)(ws + bufBytes);
  v16bf* Wp     = (v16bf*)(ws + 2 * bufBytes);                 // 32B-aligned
  float* colsum = (float*)(ws + 2 * bufBytes + (1 << 20));

  // Layer 0: 1433 -> 128   (x = in_feature)
  run_layer(X,    F_IN,  W0, b0, F_HID, arow, acol, aval, bufA, bufB, Wp, colsum, stream);
  // Layer 1: 128 -> 128    (gemm reads bufB before fill_bias rewrites it — stream-ordered)
  run_layer(bufB, F_HID, W1, b1, F_HID, arow, acol, aval, bufA, bufB, Wp, colsum, stream);
  // Layer 2: 128 -> 64
  run_layer(bufB, F_HID, W2, b2, F_OUT, arow, acol, aval, bufA, bufB, Wp, colsum, stream);

  // Decode 40000 edges -> sigmoid scores
  gcn_decode<<<(NPOS + NNEG) / 8, 256, 0, stream>>>(bufB, pos, neg, out, NPOS, NNEG, F_OUT);
}
